// MultiHeadAttention_33363305955526
// MI455X (gfx1250) — matrix operand order repaired and checked by reading, not compiled, not run on hardware
//
#include <hip/hip_runtime.h>
#include <hip/hip_bf16.h>

typedef __attribute__((ext_vector_type(16))) __bf16 v16bf;
typedef __attribute__((ext_vector_type(8)))  __bf16 v8bf;
typedef __attribute__((ext_vector_type(2)))  __bf16 v2bf;
typedef __attribute__((ext_vector_type(8)))  float  v8f;

#define S_LEN  2048
#define DMODEL 1024
#define NHEAD  16
#define DHEAD  64
#define BATCH  2
#define ROWS   (BATCH * S_LEN)     // 4096
#define BH     (BATCH * NHEAD)     // 32
#define PSTRIDE 2052               // padded LDS row stride (floats)

__device__ __forceinline__ __bf16 f2bf(float f) {
    unsigned int u = __builtin_bit_cast(unsigned int, f);
    u += 0x7FFFu + ((u >> 16) & 1u);               // round-to-nearest-even
    unsigned short h = (unsigned short)(u >> 16);
    return __builtin_bit_cast(__bf16, h);
}

// Packed f32x2 -> bf16x2: single v_cvt_pk_bf16_f32 when available.
__device__ __forceinline__ v2bf pk2bf(float x, float y) {
#if __has_builtin(__builtin_amdgcn_cvt_pk_bf16_f32)
    auto r = __builtin_amdgcn_cvt_pk_bf16_f32(x, y);
    return __builtin_bit_cast(v2bf, r);
#else
    v2bf p; p[0] = f2bf(x); p[1] = f2bf(y); return p;
#endif
}

__device__ __forceinline__ v16bf join8(v8bf lo, v8bf hi) {
    return __builtin_shufflevector(lo, hi, 0,1,2,3,4,5,6,7,8,9,10,11,12,13,14,15);
}

__device__ __forceinline__ v16bf cvt16(float4 f0, float4 f1, float4 f2, float4 f3) {
    v2bf p0 = pk2bf(f0.x, f0.y), p1 = pk2bf(f0.z, f0.w);
    v2bf p2 = pk2bf(f1.x, f1.y), p3 = pk2bf(f1.z, f1.w);
    v2bf p4 = pk2bf(f2.x, f2.y), p5 = pk2bf(f2.z, f2.w);
    v2bf p6 = pk2bf(f3.x, f3.y), p7 = pk2bf(f3.z, f3.w);
    v8bf lo = __builtin_shufflevector(__builtin_shufflevector(p0, p1, 0,1,2,3),
                                      __builtin_shufflevector(p2, p3, 0,1,2,3),
                                      0,1,2,3,4,5,6,7);
    v8bf hi = __builtin_shufflevector(__builtin_shufflevector(p4, p5, 0,1,2,3),
                                      __builtin_shufflevector(p6, p7, 0,1,2,3),
                                      0,1,2,3,4,5,6,7);
    return join8(lo, hi);
}

__device__ __forceinline__ v8f wmma_bf16(v16bf a, v16bf b, v8f c) {
    return __builtin_amdgcn_wmma_f32_16x16x32_bf16(false, a, false, b,
                                                   (short)0, c, false, false);
}

// ---------------------------------------------------------------------------
// Fragment layouts (ISA 7.12.2), half = lane>>4, ln15 = lane&15:
//  A 16x32 bf16: row = ln15; element i: K = k0 + (i<8 ? 0 : 16) + half*8 + (i&7)
//                => two contiguous 8-element (16B) loads at k0+half*8 and +16.
//  B 32x16 bf16: col = ln15; element i: K = k0 + (i<8 ? 0 : 16) + half*8 + (i&7) (same as A)
//                => two contiguous 8-element (16B) loads at k0+half*8 and +16.
//  C/D f32 16x16: vgpr i: M = i + half*8, N = ln15.
// ---------------------------------------------------------------------------

// Wt[n][k] (bf16) = W[k][n], 16x16 LDS tile transpose.
__global__ __launch_bounds__(256)
void wconvert_kernel(const float* __restrict__ W, __bf16* __restrict__ Wt) {
    __shared__ float tile[16][17];
    const int tx = threadIdx.x & 15, ty = threadIdx.x >> 4;
    const int n0 = blockIdx.x * 16, k0 = blockIdx.y * 16;
    tile[ty][tx] = W[(size_t)(k0 + ty) * DMODEL + n0 + tx];
    __syncthreads();
    Wt[(size_t)(n0 + ty) * DMODEL + k0 + tx] = f2bf(tile[tx][ty]);
}

// Y = X @ W + bias. MODE 0: Y[bh][s][d] (head split); MODE 1: Y[bh][d][s] (V^T).
template <int MODE>
__global__ __launch_bounds__(32)
void proj_kernel(const float* __restrict__ X, const __bf16* __restrict__ Wt,
                 const float* __restrict__ bias, __bf16* __restrict__ Y) {
    const int lane = threadIdx.x;
    const int half = lane >> 4, ln15 = lane & 15;
    const int m0 = blockIdx.y * 16;
    const int n0 = blockIdx.x * 64;
    const float* Xrow = X + (size_t)(m0 + ln15) * DMODEL;
    v8f acc[4] = {};
    for (int k0 = 0; k0 < DMODEL; k0 += 32) {
        const float* p = Xrow + k0 + half * 8;
        v16bf a = cvt16(*(const float4*)(p),      *(const float4*)(p + 4),
                        *(const float4*)(p + 16), *(const float4*)(p + 20));
#pragma unroll
        for (int ns = 0; ns < 4; ++ns) {
            const __bf16* wp = Wt + (size_t)(n0 + ns * 16 + ln15) * DMODEL
                                  + k0 + half * 8;
            v16bf b = join8(*(const v8bf*)wp, *(const v8bf*)(wp + 16));
            acc[ns] = wmma_bf16(a, b, acc[ns]);
        }
    }
#pragma unroll
    for (int ns = 0; ns < 4; ++ns)
#pragma unroll
        for (int i = 0; i < 8; ++i) {
            int row = m0 + i + half * 8;
            int col = n0 + ns * 16 + ln15;
            float val = acc[ns][i] + bias[col];
            int h = col >> 6, d = col & 63;
            int bb = row >> 11, s = row & (S_LEN - 1);
            if (MODE == 0)
                Y[((size_t)(bb * NHEAD + h) * S_LEN + s) * DHEAD + d] = f2bf(val);
            else
                Y[((size_t)(bb * NHEAD + h) * DHEAD + d) * S_LEN + s] = f2bf(val);
        }
}

// Fused: scores -> LDS, softmax in LDS, write P once, O = P@V from LDS.
// One workgroup (8 waves) per (bh, 16-query-row block).
__global__ __launch_bounds__(256)
void attn_kernel(const __bf16* __restrict__ Qbf, const __bf16* __restrict__ Kbf,
                 const __bf16* __restrict__ Vt, __bf16* __restrict__ Obf,
                 float* __restrict__ attn) {
    __shared__ float Pbuf[16 * PSTRIDE];   // ~128 KB (ISA: up to 320KB per WG)
    __shared__ float Obuf[16 * DHEAD];     // 4 KB
    __shared__ float red[16 * 17];
    const int t = threadIdx.x;
    const int lane = t & 31, w = t >> 5;
    const int half = lane >> 4, ln15 = lane & 15;
    const int bh = blockIdx.y;
    const int m0 = blockIdx.x * 16;

    const __bf16* Q = Qbf + (size_t)bh * S_LEN * DHEAD;
    const __bf16* K = Kbf + (size_t)bh * S_LEN * DHEAD;
    const __bf16* V = Vt  + (size_t)bh * DHEAD * S_LEN;

    for (int idx = t; idx < 16 * DHEAD; idx += 256) Obuf[idx] = 0.0f;

    // ---- scores: S = (Q K^T) * 0.125 into LDS ----
    const __bf16* qrow = Q + (size_t)(m0 + ln15) * DHEAD + half * 8;
    v16bf a0 = join8(*(const v8bf*)(qrow),      *(const v8bf*)(qrow + 16));
    v16bf a1 = join8(*(const v8bf*)(qrow + 32), *(const v8bf*)(qrow + 48));
    for (int j = 0; j < 16; ++j) {
        int n0 = (w + 8 * j) * 16;                  // 8 waves cover 128 n-tiles
        const __bf16* krow = K + (size_t)(n0 + ln15) * DHEAD + half * 8;
        if (j < 15)                                  // next tile for this wave
            __builtin_prefetch(krow + 128 * DHEAD, 0, 3);
        v16bf b0 = join8(*(const v8bf*)(krow),      *(const v8bf*)(krow + 16));
        v16bf b1 = join8(*(const v8bf*)(krow + 32), *(const v8bf*)(krow + 48));
        v8f s = {};
        s = wmma_bf16(a0, b0, s);
        s = wmma_bf16(a1, b1, s);
#pragma unroll
        for (int i = 0; i < 8; ++i)
            Pbuf[(i + half * 8) * PSTRIDE + n0 + ln15] = s[i] * 0.125f;
    }
    __syncthreads();

    // ---- softmax over each of the 16 rows (full 2048 cols in LDS) ----
    {
        const int r = t >> 4, cs = t & 15;
        float mx = -INFINITY;
        for (int c = cs; c < S_LEN; c += 16)
            mx = fmaxf(mx, Pbuf[r * PSTRIDE + c]);
        red[r * 17 + cs] = mx;
        __syncthreads();
        if (cs == 0) {
            float m2 = red[r * 17];
            for (int i = 1; i < 16; ++i) m2 = fmaxf(m2, red[r * 17 + i]);
            red[r * 17 + 16] = m2;
        }
        __syncthreads();
        const float rmax = red[r * 17 + 16];
        float sum = 0.0f;
        for (int c = cs; c < S_LEN; c += 16) {
            float e = __expf(Pbuf[r * PSTRIDE + c] - rmax);
            Pbuf[r * PSTRIDE + c] = e;
            sum += e;
        }
        __syncthreads();
        red[r * 17 + cs] = sum;
        __syncthreads();
        if (cs == 0) {
            float s2 = 0.0f;
            for (int i = 0; i < 16; ++i) s2 += red[r * 17 + i];
            red[r * 17 + 16] = 1.0f / s2;
        }
        __syncthreads();
    }
    // normalize in LDS and stream P to global exactly once (coalesced)
    float* Pg = attn + (size_t)bh * S_LEN * S_LEN + (size_t)m0 * S_LEN;
    for (int idx = t; idx < 16 * S_LEN; idx += 256) {
        int r = idx >> 11, c = idx & (S_LEN - 1);
        float val = Pbuf[r * PSTRIDE + c] * red[r * 17 + 16];
        Pbuf[r * PSTRIDE + c] = val;
        Pg[(size_t)r * S_LEN + c] = val;
    }
    __syncthreads();

    // ---- O = P @ V from LDS; waves split K range, ds_add combine ----
    {
        v8f acc[4] = {};
        const int kb0 = w * 256;
        for (int kk = 0; kk < 256; kk += 32) {
            const int k0 = kb0 + kk;
            const float* prow = Pbuf + ln15 * PSTRIDE + k0 + half * 8;
            v16bf a = cvt16(*(const float4*)(prow),      *(const float4*)(prow + 4),
                            *(const float4*)(prow + 16), *(const float4*)(prow + 20));
#pragma unroll
            for (int ns = 0; ns < 4; ++ns) {
                const __bf16* vp = V + (size_t)(ns * 16 + ln15) * S_LEN
                                     + k0 + half * 8;
                if (ns == 0 && kk < 224)
                    __builtin_prefetch(vp + 32, 0, 3);
                v16bf b = join8(*(const v8bf*)vp, *(const v8bf*)(vp + 16));
                acc[ns] = wmma_bf16(a, b, acc[ns]);
            }
        }
#pragma unroll
        for (int ns = 0; ns < 4; ++ns)
#pragma unroll
            for (int i = 0; i < 8; ++i)
                atomicAdd(&Obuf[(i + half * 8) * DHEAD + ns * 16 + ln15],
                          acc[ns][i]);
    }
    __syncthreads();
    __bf16* Og = Obf + ((size_t)bh * S_LEN + m0) * DHEAD;
    for (int idx = t; idx < 16 * DHEAD; idx += 256)
        Og[idx] = f2bf(Obuf[idx]);
}

// out = merged(attn_out) @ Wo + bo (head merge folded into A-fragment loads).
__global__ __launch_bounds__(32)
void out_kernel(const __bf16* __restrict__ Obf, const __bf16* __restrict__ WtO,
                const float* __restrict__ bo, float* __restrict__ out) {
    const int lane = threadIdx.x;
    const int half = lane >> 4, ln15 = lane & 15;
    const int m0 = blockIdx.y * 16;
    const int n0 = blockIdx.x * 64;
    const int row = m0 + ln15, bb = row >> 11, s = row & (S_LEN - 1);
    v8f acc[4] = {};
    for (int k0 = 0; k0 < DMODEL; k0 += 32) {
        const int h = k0 >> 6;                       // 32-aligned span: one head
        const int d = (k0 & 63) + half * 8;
        const __bf16* orow = Obf + ((size_t)(bb * NHEAD + h) * S_LEN + s) * DHEAD + d;
        v16bf a = join8(*(const v8bf*)orow, *(const v8bf*)(orow + 16));
#pragma unroll
        for (int ns = 0; ns < 4; ++ns) {
            const __bf16* wp = WtO + (size_t)(n0 + ns * 16 + ln15) * DMODEL
                                   + k0 + half * 8;
            v16bf b = join8(*(const v8bf*)wp, *(const v8bf*)(wp + 16));
            acc[ns] = wmma_bf16(a, b, acc[ns]);
        }
    }
#pragma unroll
    for (int ns = 0; ns < 4; ++ns)
#pragma unroll
        for (int i = 0; i < 8; ++i) {
            int r = m0 + i + half * 8;
            int col = n0 + ns * 16 + ln15;
            out[(size_t)r * DMODEL + col] = acc[ns][i] + bo[col];
        }
}

extern "C" void kernel_launch(void* const* d_in, const int* in_sizes, int n_in,
                              void* d_out, int out_size, void* d_ws, size_t ws_size,
                              hipStream_t stream) {
    (void)in_sizes; (void)n_in; (void)out_size; (void)ws_size;
    const float* q  = (const float*)d_in[0];
    const float* k  = (const float*)d_in[1];
    const float* v  = (const float*)d_in[2];
    /* d_in[3] = mask: reference does NOT apply it */
    const float* Wq = (const float*)d_in[4];
    const float* bq = (const float*)d_in[5];
    const float* Wk = (const float*)d_in[6];
    const float* bk = (const float*)d_in[7];
    const float* Wv = (const float*)d_in[8];
    const float* bv = (const float*)d_in[9];
    const float* Wo = (const float*)d_in[10];
    const float* bo = (const float*)d_in[11];

    float* out  = (float*)d_out;
    float* attn = out + (size_t)ROWS * DMODEL;

    __bf16* Qbf  = (__bf16*)d_ws;                        // 8 MB
    __bf16* Kbf  = Qbf  + (size_t)ROWS * DMODEL;         // 8 MB
    __bf16* Vtbf = Kbf  + (size_t)ROWS * DMODEL;         // 8 MB  [bh][d][s]
    __bf16* Obf  = Vtbf + (size_t)ROWS * DMODEL;         // 8 MB  [bh][s][d]
    __bf16* WtQ  = Obf  + (size_t)ROWS * DMODEL;         // 2 MB each
    __bf16* WtK  = WtQ  + (size_t)DMODEL * DMODEL;
    __bf16* WtV  = WtK  + (size_t)DMODEL * DMODEL;
    __bf16* WtO  = WtV  + (size_t)DMODEL * DMODEL;

    dim3 tb(256), tg(DMODEL / 16, DMODEL / 16);
    wconvert_kernel<<<tg, tb, 0, stream>>>(Wq, WtQ);
    wconvert_kernel<<<tg, tb, 0, stream>>>(Wk, WtK);
    wconvert_kernel<<<tg, tb, 0, stream>>>(Wv, WtV);
    wconvert_kernel<<<tg, tb, 0, stream>>>(Wo, WtO);

    dim3 wave(32);
    dim3 projGrid(DMODEL / 64, ROWS / 16);               // 16 x 256
    proj_kernel<0><<<projGrid, wave, 0, stream>>>(q, WtQ, bq, Qbf);
    proj_kernel<0><<<projGrid, wave, 0, stream>>>(k, WtK, bk, Kbf);
    proj_kernel<1><<<projGrid, wave, 0, stream>>>(v, WtV, bv, Vtbf);

    attn_kernel<<<dim3(S_LEN / 16, BH), dim3(256), 0, stream>>>(Qbf, Kbf, Vtbf,
                                                                Obf, attn);

    out_kernel<<<projGrid, wave, 0, stream>>>(Obf, WtO, bo, out);
}
